// GraphMambaGMN_63067299775141
// MI455X (gfx1250) — matrix operand order repaired and checked by reading, not compiled, not run on hardware
//
#include <hip/hip_runtime.h>
#include <hip/hip_bf16.h>

// ---------------- problem constants (from reference) ----------------
constexpr int BB   = 8;      // batch
constexpr int NNODE= 256;    // nodes
constexpr int CIN  = 64;
constexpr int DM   = 128;    // d_model
constexpr int LT   = 16;     // tokens per node (local seq len)
constexpr int KS   = 8;      // subtokens pooled
constexpr int DSTATE = 16;
constexpr int DC   = 4;      // conv width
constexpr int DI   = 256;    // d_inner
constexpr int DTR  = 8;      // dt rank
constexpr int EE   = 8192;   // edges
constexpr int SEQS = BB * NNODE;  // 2048 local sequences
constexpr int GLN  = NNODE;       // global seq len = 256

// ---------------- WMMA helpers (gfx1250 wave32) ----------------
typedef __attribute__((ext_vector_type(16))) _Float16 v16h;
typedef __attribute__((ext_vector_type(8)))  float    v8f;
typedef __attribute__((ext_vector_type(4)))  unsigned int u32x4;
typedef __attribute__((ext_vector_type(8)))  int      i32x8;
typedef __attribute__((ext_vector_type(4)))  int      i32x4;

union HFrag { v16h h; u32x4 q[2]; };

// A fragment, 16x32 f16, from row-major [16][stride] halves.
// lane: M = lane%16; halves 0..7 -> K = k0 + (lane>=16?8:0) + e; halves 8..15 -> +16
__device__ __forceinline__ v16h load_a_frag(const _Float16* base, int stride, int lane, int k0) {
  const int M   = lane & 15;
  const int klo = (lane >> 4) << 3;   // 0 or 8
  const _Float16* r = base + M * stride + k0 + klo;
  HFrag f;
  f.q[0] = *(const u32x4*)(r);
  f.q[1] = *(const u32x4*)(r + 16);
  return f.h;
}

// B fragment, 32x16 f16, weight stored row-major [N][kdim] (i.e. W, used as B = W^T logically).
// lane: N = lane%16; halves 0..7 -> K = k0 + (lane>=16?8:0) + e; halves 8..15 -> +16
__device__ __forceinline__ v16h load_b_frag(const _Float16* w, int kdim, int lane, int n0, int k0) {
  const int N   = lane & 15;
  const int khi = (lane >> 4) << 3;   // 0 or 8
  const _Float16* r = w + (size_t)(n0 + N) * kdim + k0 + khi;
  HFrag f;
  f.q[0] = *(const u32x4*)(r);
  f.q[1] = *(const u32x4*)(r + 16);
  return f.h;
}

__device__ __forceinline__ v8f wmma_f16(v16h a, v16h b, v8f c) {
  return __builtin_amdgcn_wmma_f32_16x16x32_f16(false, a, false, b, (short)0, c, false, false);
}

__device__ __forceinline__ float siluf(float v)     { return v / (1.f + __expf(-v)); }
__device__ __forceinline__ float softplusf(float v) { return (v > 20.f) ? v : __logf(1.f + __expf(v)); }

// LDS byte address of a generic pointer into static shared memory
__device__ __forceinline__ unsigned lds_addr_of(const void* p) {
  return (unsigned)(unsigned long long)(__attribute__((address_space(3))) const char*)p;
}

// TDM: async-copy `n4B` dwords from global `src` into LDS at byte addr `lds_b`.
// D# per CDNA5 ISA: 1D tile (tile_dim0 = n4B elems of 4B),
// groups 2/3 zero (<=2D tensor). Issued once per calling wave; TENSORcnt tracks it.
// 6-arg builtin form (clang-23 / therock-10.0): (v4u, v8i, v4i, v4i, v8i, i32 cpol).
__device__ __forceinline__ void tdm_load_1d(const void* src, unsigned lds_b, unsigned n4B) {
  unsigned long long ga = (unsigned long long)src;
  u32x4 g0;
  g0[0] = 1u;                                            // count=1, user desc
  g0[1] = lds_b;                                         // lds_addr (bytes)
  g0[2] = (unsigned)(ga & 0xFFFFFFFFull);                // global_addr[31:0]
  g0[3] = (unsigned)((ga >> 32) & 0x1FFFFFFull)          // global_addr[56:32]
        | 0x80000000u;                                   // type=2 ("image")
  i32x8 g1;
  g1[0] = (int)0x00020000u;                              // data_size=2 (4 bytes)
  g1[1] = (int)((n4B & 0xFFFFu) << 16);                  // tensor_dim0[15:0] @ bits 63:48
  g1[2] = (int)(((n4B >> 16) & 0xFFFFu)                  // tensor_dim0[31:16]
              | (1u << 16));                             // tensor_dim1 = 1
  g1[3] = (int)((n4B & 0xFFFFu) << 16);                  // tile_dim0 @ bits 127:112
  g1[4] = 1;                                             // tile_dim1 = 1
  g1[5] = (int)n4B;                                      // tensor_dim0_stride[31:0]
  g1[6] = (int)((n4B & 0xFFFFu) << 16);                  // tensor_dim1_stride[15:0]
  g1[7] = 0;
  i32x4 g2 = {0, 0, 0, 0};
  i32x4 g3 = {0, 0, 0, 0};
  i32x8 gz = {0, 0, 0, 0, 0, 0, 0, 0};
  __builtin_amdgcn_tensor_load_to_lds(g0, g1, g2, g3, gz, 0);
}

// ---------------- weight conversion ----------------
__global__ void k_f32_to_f16(const float* __restrict__ s, _Float16* __restrict__ d, int n) {
  int i = blockIdx.x * 256 + threadIdx.x;
  if (i < n) d[i] = (_Float16)s[i];
}

// xproj (8,40,256) -> padded (8,48,256) f16, zero rows 40..47
__global__ void k_xproj_pad(const float* __restrict__ s, _Float16* __restrict__ d) {
  int i = blockIdx.x * 256 + threadIdx.x;
  if (i >= 8 * 48 * DI) return;
  int k = i % DI; int n = (i / DI) % 48; int p = i / (48 * DI);
  d[i] = (n < 40) ? (_Float16)s[((p * 40 + n) * DI) + k] : (_Float16)0.f;
}

// ---------------- encoder: pooling + 2-layer MLP (WMMA) ----------------
__global__ void __launch_bounds__(256) k_encoder(
    const float* __restrict__ x, const int* __restrict__ tok,
    const _Float16* __restrict__ w1h, const float* __restrict__ b1,
    const _Float16* __restrict__ w2h, const float* __restrict__ b2,
    float* __restrict__ h0) {
  __shared__ __align__(16) _Float16 sP[LT * CIN];
  __shared__ __align__(16) _Float16 sH[LT * DM];
  const int seq = blockIdx.x;
  const int b = seq / NNODE, n = seq % NNODE;
  const int tid = threadIdx.x, lane = tid & 31, wave = tid >> 5;

  for (int cell = tid; cell < LT * CIN; cell += 256) {
    int t = cell / CIN, c = cell % CIN;
    float s = 0.f;
    for (int k = 0; k < KS; ++k) {
      int id = tok[(n * LT + t) * KS + k];
      s += x[(b * NNODE + id) * CIN + c];
    }
    sP[t * CIN + c] = (_Float16)(s * (1.f / KS));
  }
  __syncthreads();
  {
    int n0 = wave * 16;
    v8f acc = {};
    for (int k0 = 0; k0 < CIN; k0 += 32)
      acc = wmma_f16(load_a_frag(sP, CIN, lane, k0), load_b_frag(w1h, CIN, lane, n0, k0), acc);
    int col = lane & 15, hi = (lane >> 4) * 8;
    for (int j = 0; j < 8; ++j) {
      int row = j + hi, nn = n0 + col;
      sH[row * DM + nn] = (_Float16)fmaxf(acc[j] + b1[nn], 0.f);
    }
  }
  __syncthreads();
  {
    int n0 = wave * 16;
    v8f acc = {};
    for (int k0 = 0; k0 < DM; k0 += 32)
      acc = wmma_f16(load_a_frag(sH, DM, lane, k0), load_b_frag(w2h, DM, lane, n0, k0), acc);
    int col = lane & 15, hi = (lane >> 4) * 8;
    for (int j = 0; j < 8; ++j) {
      int row = j + hi, nn = n0 + col;
      h0[((size_t)seq * LT + row) * DM + nn] = acc[j] + b2[nn];
    }
  }
}

// ---------------- fused local bimamba (L=16, one seq per block) ----------------
// LDS layout (54 KB, staged reuse):
//  [0     ,16384) XH f32 16x256 (in_proj->conv), reused as DT f32 (dt->scan)
//  [16384 ,24576) XCh f16 16x256 (conv->scan)
//  [24576 ,32768) Zh  f16 16x256 (in_proj->scan)
//  [32768 ,40960) DTW f32 256x8 (TDM load -> dt stage), reused as Yh f16 16x256 (scan->out_proj)
//  [40960 ,49152) Acc f32 16x128 (whole kernel)
//  [49152 ,53248) A   f16 16x128 (load->in_proj), reused as DBC f32 16x48
//  [53248 ,55296) Bm,Cm f32 16x16 each
__global__ void __launch_bounds__(256) k_local_bimamba(
    const float* __restrict__ h_in, float* __restrict__ h_out, int p0,
    const _Float16* __restrict__ in_w_h, const float* __restrict__ conv_w,
    const float* __restrict__ conv_b, const _Float16* __restrict__ xproj_h,
    const float* __restrict__ dt_w, const float* __restrict__ dt_b,
    const float* __restrict__ Alog, const float* __restrict__ Dp,
    const _Float16* __restrict__ out_w_h) {
  __shared__ __align__(16) char smem[55296];
  float*    sXH  = (float*)(smem + 0);
  float*    sDT  = (float*)(smem + 0);
  _Float16* sXCh = (_Float16*)(smem + 16384);
  _Float16* sZh  = (_Float16*)(smem + 24576);
  float*    sDTW = (float*)(smem + 32768);   // dt_w block via TDM (dies before scan)
  _Float16* sYh  = (_Float16*)(smem + 32768);
  float*    sAcc = (float*)(smem + 40960);
  _Float16* sA   = (_Float16*)(smem + 49152);
  float*    sDBC = (float*)(smem + 49152);
  float*    sBm  = (float*)(smem + 53248);
  float*    sCm  = (float*)(smem + 54272);

  const int seq = blockIdx.x;
  const int tid = threadIdx.x, lane = tid & 31, wave = tid >> 5;

  for (int i = tid; i < LT * DM; i += 256) sAcc[i] = 0.f;
  __syncthreads();

  for (int dir = 0; dir < 2; ++dir) {
    const int p = p0 + dir;
    // kick off the TDM copy of this direction's dt_w block (256x8 f32 = 8KB)
    // into LDS; it overlaps in_proj/conv/xproj and is awaited before the dt stage.
    if (wave == 0)
      tdm_load_1d(dt_w + (size_t)p * DI * DTR, lds_addr_of(sDTW), DI * DTR);

    for (int i = tid; i < LT * DM; i += 256) {
      int r = i / DM, c = i % DM;
      int rs = dir ? (LT - 1 - r) : r;
      sA[r * DM + c] = (_Float16)h_in[((size_t)seq * LT + rs) * DM + c];
    }
    __syncthreads();
    // in_proj: 16x128 @ (512x128)^T -> xh(16x256), z(16x256)
    const _Float16* Win = in_w_h + (size_t)p * 512 * DM;
    for (int tt = 0; tt < 4; ++tt) {
      int n0 = (wave * 4 + tt) * 16;
      v8f acc = {};
      for (int k0 = 0; k0 < DM; k0 += 32)
        acc = wmma_f16(load_a_frag(sA, DM, lane, k0), load_b_frag(Win, DM, lane, n0, k0), acc);
      int col = lane & 15, hi = (lane >> 4) * 8;
      for (int j = 0; j < 8; ++j) {
        int row = j + hi, nn = n0 + col;
        if (nn < DI) sXH[row * DI + nn] = acc[j];
        else         sZh[row * DI + (nn - DI)] = (_Float16)acc[j];
      }
    }
    __syncthreads();
    // causal conv + silu (thread = channel)
    {
      const int d = tid;
      float cw[DC];
      for (int j = 0; j < DC; ++j) cw[j] = conv_w[(p * DI + d) * DC + j];
      float cb = conv_b[p * DI + d];
      for (int t = 0; t < LT; ++t) {
        float s = cb;
        for (int j = 0; j < DC; ++j) {
          int ts = t - (DC - 1) + j;
          if (ts >= 0) s += sXH[ts * DI + d] * cw[j];
        }
        sXCh[t * DI + d] = (_Float16)siluf(s);
      }
    }
    __syncthreads();
    // xproj: 16x256 @ (48x256)^T -> dbc (waves 0..2)
    const _Float16* Wxp = xproj_h + (size_t)p * 48 * DI;
    if (wave < 3) {
      int n0 = wave * 16;
      v8f acc = {};
      for (int k0 = 0; k0 < DI; k0 += 32)
        acc = wmma_f16(load_a_frag(sXCh, DI, lane, k0), load_b_frag(Wxp, DI, lane, n0, k0), acc);
      int col = lane & 15, hi = (lane >> 4) * 8;
      for (int j = 0; j < 8; ++j) sDBC[(j + hi) * 48 + n0 + col] = acc[j];
    }
    // publish the TDM-loaded dt_w block to the workgroup
    if (wave == 0) __builtin_amdgcn_s_wait_tensorcnt(0);
    __syncthreads();
    // dt = softplus(dbc[:, :8] @ dt_w^T + b); split Bm/Cm
    {
      const int d = tid;
      const float* dw = sDTW + d * DTR;   // from LDS (TDM)
      float db = dt_b[p * DI + d];
      for (int t = 0; t < LT; ++t) {
        float s = db;
        for (int r = 0; r < DTR; ++r) s += sDBC[t * 48 + r] * dw[r];
        sDT[t * DI + d] = softplusf(s);
      }
      int t = tid >> 4, si = tid & 15;   // 256 threads == 16*16
      sBm[t * DSTATE + si] = sDBC[t * 48 + DTR + si];
      sCm[t * DSTATE + si] = sDBC[t * 48 + DTR + DSTATE + si];
    }
    __syncthreads();
    // selective scan (thread = channel, state in regs)
    {
      const int d = tid;
      float As[DSTATE], hs[DSTATE];
      for (int s = 0; s < DSTATE; ++s) { As[s] = -__expf(Alog[(p * DI + d) * DSTATE + s]); hs[s] = 0.f; }
      float dpar = Dp[p * DI + d];
      for (int t = 0; t < LT; ++t) {
        float dtv = sDT[t * DI + d];
        float xv  = (float)sXCh[t * DI + d];
        float dx  = dtv * xv;
        float y = 0.f;
        #pragma unroll
        for (int s = 0; s < DSTATE; ++s) {
          float e = __expf(dtv * As[s]);
          hs[s] = e * hs[s] + dx * sBm[t * DSTATE + s];
          y += hs[s] * sCm[t * DSTATE + s];
        }
        float zv = (float)sZh[t * DI + d];
        sYh[t * DI + d] = (_Float16)((y + xv * dpar) * siluf(zv));
      }
    }
    __syncthreads();
    // out_proj: 16x256 @ (128x256)^T, accumulate 0.5x with flip
    {
      const _Float16* Wout = out_w_h + (size_t)p * DM * DI;
      int n0 = wave * 16;
      v8f acc = {};
      for (int k0 = 0; k0 < DI; k0 += 32)
        acc = wmma_f16(load_a_frag(sYh, DI, lane, k0), load_b_frag(Wout, DI, lane, n0, k0), acc);
      int col = lane & 15, hi = (lane >> 4) * 8;
      for (int j = 0; j < 8; ++j) {
        int row = j + hi, nn = n0 + col;
        int ro = dir ? (LT - 1 - row) : row;
        sAcc[ro * DM + nn] += 0.5f * acc[j];
      }
    }
    __syncthreads();
  }
  for (int i = tid; i < LT * DM; i += 256)
    h_out[(size_t)seq * LT * DM + i] = sAcc[i];
}

// ---------------- global-layer pipeline (L=256) ----------------
__global__ void __launch_bounds__(256) k_g_inproj(
    const float* __restrict__ hg_in, float* __restrict__ xh_g, _Float16* __restrict__ zh_g,
    int p, int dir, const _Float16* __restrict__ in_w_h) {
  __shared__ __align__(16) _Float16 sA[16 * DM];
  const int bx = blockIdx.x;                 // 8 seq * 16 mtiles
  const int seq = bx >> 4, t0 = (bx & 15) * 16;
  const int tid = threadIdx.x, lane = tid & 31, wave = tid >> 5;
  for (int i = tid; i < 16 * DM; i += 256) {
    int r = i / DM, c = i % DM;
    int rs = dir ? (GLN - 1 - (t0 + r)) : (t0 + r);
    sA[i] = (_Float16)hg_in[((size_t)seq * GLN + rs) * DM + c];
  }
  __syncthreads();
  const _Float16* Win = in_w_h + (size_t)p * 512 * DM;
  for (int tt = 0; tt < 4; ++tt) {
    int n0 = (wave * 4 + tt) * 16;
    v8f acc = {};
    for (int k0 = 0; k0 < DM; k0 += 32)
      acc = wmma_f16(load_a_frag(sA, DM, lane, k0), load_b_frag(Win, DM, lane, n0, k0), acc);
    int col = lane & 15, hi = (lane >> 4) * 8;
    for (int j = 0; j < 8; ++j) {
      int row = j + hi, nn = n0 + col;
      size_t g = (size_t)seq * GLN + t0 + row;
      if (nn < DI) xh_g[g * DI + nn] = acc[j];
      else         zh_g[g * DI + (nn - DI)] = (_Float16)acc[j];
    }
  }
}

__global__ void k_g_conv(const float* __restrict__ xh_g, _Float16* __restrict__ xch_g,
                         int p, const float* __restrict__ conv_w, const float* __restrict__ conv_b) {
  int idx = blockIdx.x * 256 + threadIdx.x;  // BB*GLN*DI
  if (idx >= BB * GLN * DI) return;
  int d = idx % DI; int t = (idx / DI) % GLN; int seq = idx / (DI * GLN);
  float s = conv_b[p * DI + d];
  for (int j = 0; j < DC; ++j) {
    int ts = t - (DC - 1) + j;
    if (ts >= 0) s += xh_g[((size_t)seq * GLN + ts) * DI + d] * conv_w[(p * DI + d) * DC + j];
  }
  xch_g[idx] = (_Float16)siluf(s);
}

__global__ void __launch_bounds__(256) k_g_xproj(
    const _Float16* __restrict__ xch_g, float* __restrict__ dt_g,
    float* __restrict__ Bm_g, float* __restrict__ Cm_g,
    int p, const _Float16* __restrict__ xproj_h,
    const float* __restrict__ dt_w, const float* __restrict__ dt_b) {
  __shared__ __align__(16) float sDBC[16 * 48];
  const int bx = blockIdx.x; int seq = bx >> 4, t0 = (bx & 15) * 16;
  const int tid = threadIdx.x, lane = tid & 31, wave = tid >> 5;
  const _Float16* Abase = xch_g + ((size_t)seq * GLN + t0) * DI;
  const _Float16* Wxp = xproj_h + (size_t)p * 48 * DI;
  if (wave < 3) {
    int n0 = wave * 16;
    v8f acc = {};
    for (int k0 = 0; k0 < DI; k0 += 32)
      acc = wmma_f16(load_a_frag(Abase, DI, lane, k0), load_b_frag(Wxp, DI, lane, n0, k0), acc);
    int col = lane & 15, hi = (lane >> 4) * 8;
    for (int j = 0; j < 8; ++j) sDBC[(j + hi) * 48 + n0 + col] = acc[j];
  }
  __syncthreads();
  {
    const int d = tid;
    const float* dw = dt_w + (p * DI + d) * DTR;
    float db = dt_b[p * DI + d];
    for (int t = 0; t < 16; ++t) {
      float s = db;
      for (int r = 0; r < DTR; ++r) s += sDBC[t * 48 + r] * dw[r];
      dt_g[((size_t)seq * GLN + t0 + t) * DI + d] = softplusf(s);
    }
    int t = tid >> 4, si = tid & 15;
    Bm_g[((size_t)seq * GLN + t0 + t) * DSTATE + si] = sDBC[t * 48 + DTR + si];
    Cm_g[((size_t)seq * GLN + t0 + t) * DSTATE + si] = sDBC[t * 48 + DTR + DSTATE + si];
  }
}

__global__ void __launch_bounds__(256) k_g_scan(
    const float* __restrict__ dt_g, const _Float16* __restrict__ xch_g,
    const _Float16* __restrict__ zh_g,
    const float* __restrict__ Bm_g, const float* __restrict__ Cm_g,
    _Float16* __restrict__ yh_g, int p,
    const float* __restrict__ Alog, const float* __restrict__ Dp) {
  const int seq = blockIdx.x;
  const int d = threadIdx.x;
  float As[DSTATE], hs[DSTATE];
  for (int s = 0; s < DSTATE; ++s) { As[s] = -__expf(Alog[(p * DI + d) * DSTATE + s]); hs[s] = 0.f; }
  float dpar = Dp[p * DI + d];
  for (int t = 0; t < GLN; ++t) {
    size_t g = (size_t)seq * GLN + t;
    if (t + 1 < GLN) __builtin_prefetch(&dt_g[(g + 1) * DI + d], 0, 0);
    float dtv = dt_g[g * DI + d];
    float xv  = (float)xch_g[g * DI + d];
    float dx  = dtv * xv;
    float y = 0.f;
    #pragma unroll
    for (int s = 0; s < DSTATE; ++s) {
      float e = __expf(dtv * As[s]);
      hs[s] = e * hs[s] + dx * Bm_g[g * DSTATE + s];
      y += hs[s] * Cm_g[g * DSTATE + s];
    }
    float zv = (float)zh_g[g * DI + d];
    yh_g[g * DI + d] = (_Float16)((y + xv * dpar) * siluf(zv));
  }
}

__global__ void __launch_bounds__(256) k_g_outproj(
    const _Float16* __restrict__ yh_g, float* __restrict__ hg_out,
    int p, int dir, const _Float16* __restrict__ out_w_h) {
  const int bx = blockIdx.x; int seq = bx >> 4, t0 = (bx & 15) * 16;
  const int tid = threadIdx.x, lane = tid & 31, wave = tid >> 5;
  const _Float16* Abase = yh_g + ((size_t)seq * GLN + t0) * DI;
  const _Float16* Wout = out_w_h + (size_t)p * DM * DI;
  int n0 = wave * 16;
  v8f acc = {};
  for (int k0 = 0; k0 < DI; k0 += 32)
    acc = wmma_f16(load_a_frag(Abase, DI, lane, k0), load_b_frag(Wout, DI, lane, n0, k0), acc);
  int col = lane & 15, hi = (lane >> 4) * 8;
  for (int j = 0; j < 8; ++j) {
    int row = j + hi, nn = n0 + col;
    int rs = dir ? (GLN - 1 - (t0 + row)) : (t0 + row);
    float v = 0.5f * acc[j];
    float* o = &hg_out[((size_t)seq * GLN + rs) * DM + nn];
    if (dir == 0) *o = v; else *o += v;
  }
}

// ---------------- glue: permute / inverse-permute / edges / MPNN ----------------
__global__ void k_node_perm(const float* __restrict__ hloc, const int* __restrict__ perm,
                            float* __restrict__ hg0) {
  int idx = blockIdx.x * 256 + threadIdx.x;  // BB*NNODE*DM
  if (idx >= BB * NNODE * DM) return;
  int c = idx % DM; int j = (idx / DM) % NNODE; int b = idx / (DM * NNODE);
  int pn = perm[j];
  hg0[idx] = hloc[(((size_t)b * NNODE + pn) * LT + (LT - 1)) * DM + c];
}

__global__ void k_invperm(const float* __restrict__ hg, const int* __restrict__ perm,
                          float* __restrict__ feat, float* __restrict__ nei) {
  int idx = blockIdx.x * 256 + threadIdx.x;
  if (idx >= BB * NNODE * DM) return;
  int c = idx % DM; int j = (idx / DM) % NNODE; int b = idx / (DM * NNODE);
  feat[((size_t)b * NNODE + perm[j]) * DM + c] = hg[idx];
  nei[idx] = 0.f;
}

__global__ void k_edges(const float* __restrict__ feat, float* __restrict__ nei,
                        const int* __restrict__ ei) {
  int idx = blockIdx.x * 256 + threadIdx.x;  // 2*EE*BB*DM = 16.7M
  if (idx >= 2 * EE * BB * DM) return;
  int c = idx % DM; int b = (idx / DM) % BB; int r = idx / (DM * BB);
  int e = r % EE; int dr = r / EE;
  int s0 = ei[e], d0 = ei[EE + e];
  int src = dr ? d0 : s0;
  int dst = dr ? s0 : d0;
  atomicAdd(&nei[((size_t)b * NNODE + dst) * DM + c], feat[((size_t)b * NNODE + src) * DM + c]);
}

__global__ void __launch_bounds__(256) k_mpnn(
    const float* __restrict__ feat, const float* __restrict__ nei,
    const _Float16* __restrict__ selfw, const float* __restrict__ selfb,
    const _Float16* __restrict__ neigw, const float* __restrict__ neigb,
    float* __restrict__ out) {
  __shared__ __align__(16) _Float16 sF[16 * DM];
  __shared__ __align__(16) _Float16 sN[16 * DM];
  const int r0 = blockIdx.x * 16;
  const int tid = threadIdx.x, lane = tid & 31, wave = tid >> 5;
  for (int i = tid; i < 16 * DM; i += 256) {
    sF[i] = (_Float16)feat[(size_t)r0 * DM + i];
    sN[i] = (_Float16)nei[(size_t)r0 * DM + i];
  }
  __syncthreads();
  int n0 = wave * 16;
  v8f acc = {};
  for (int k0 = 0; k0 < DM; k0 += 32) {
    acc = wmma_f16(load_a_frag(sF, DM, lane, k0), load_b_frag(selfw, DM, lane, n0, k0), acc);
    acc = wmma_f16(load_a_frag(sN, DM, lane, k0), load_b_frag(neigw, DM, lane, n0, k0), acc);
  }
  int col = lane & 15, hi = (lane >> 4) * 8;
  for (int j = 0; j < 8; ++j) {
    int row = j + hi, nn = n0 + col;
    float v = acc[j] + selfb[nn] + neigb[nn];
    float f = feat[(size_t)(r0 + row) * DM + nn];
    out[(size_t)(r0 + row) * DM + nn] = f + fmaxf(v, 0.f);
  }
}

// ---------------- host launcher ----------------
extern "C" void kernel_launch(void* const* d_in, const int* in_sizes, int n_in,
                              void* d_out, int out_size, void* d_ws, size_t ws_size,
                              hipStream_t stream) {
  const float* x         = (const float*)d_in[0];
  const float* enc_w1    = (const float*)d_in[1];
  const float* enc_b1    = (const float*)d_in[2];
  const float* enc_w2    = (const float*)d_in[3];
  const float* enc_b2    = (const float*)d_in[4];
  const float* m_in_w    = (const float*)d_in[5];
  const float* m_conv_w  = (const float*)d_in[6];
  const float* m_conv_b  = (const float*)d_in[7];
  const float* m_xproj_w = (const float*)d_in[8];
  const float* m_dt_w    = (const float*)d_in[9];
  const float* m_dt_b    = (const float*)d_in[10];
  const float* m_Alog    = (const float*)d_in[11];
  const float* m_D       = (const float*)d_in[12];
  const float* m_out_w   = (const float*)d_in[13];
  const float* mp_self_w = (const float*)d_in[14];
  const float* mp_self_b = (const float*)d_in[15];
  const float* mp_neig_w = (const float*)d_in[16];
  const float* mp_neig_b = (const float*)d_in[17];
  const int*   token_ids = (const int*)d_in[18];
  const int*   edge_index= (const int*)d_in[19];
  const int*   perm      = (const int*)d_in[20];
  float* out = (float*)d_out;
  (void)in_sizes; (void)n_in; (void)out_size; (void)ws_size;

  char* ws = (char*)d_ws;
  size_t o = 0;
  auto take = [&](size_t bytes) -> char* {
    char* r = ws + o; o = (o + bytes + 255) & ~(size_t)255; return r;
  };
  float*     hA    = (float*)take((size_t)SEQS * LT * DM * 4);   // 16 MB
  float*     hB    = (float*)take((size_t)SEQS * LT * DM * 4);   // 16 MB
  float*     hgA   = (float*)take((size_t)BB * NNODE * DM * 4);
  float*     hgB   = (float*)take((size_t)BB * NNODE * DM * 4);
  float*     feat  = (float*)take((size_t)BB * NNODE * DM * 4);
  float*     nei   = (float*)take((size_t)BB * NNODE * DM * 4);
  float*     xh_g  = (float*)take((size_t)BB * GLN * DI * 4);
  float*     dt_g  = (float*)take((size_t)BB * GLN * DI * 4);
  float*     Bm_g  = (float*)take((size_t)BB * GLN * DSTATE * 4);
  float*     Cm_g  = (float*)take((size_t)BB * GLN * DSTATE * 4);
  _Float16*  zh_g  = (_Float16*)take((size_t)BB * GLN * DI * 2);
  _Float16*  xch_g = (_Float16*)take((size_t)BB * GLN * DI * 2);
  _Float16*  yh_g  = (_Float16*)take((size_t)BB * GLN * DI * 2);
  _Float16*  enc1h = (_Float16*)take((size_t)DM * CIN * 2);
  _Float16*  enc2h = (_Float16*)take((size_t)DM * DM * 2);
  _Float16*  inwh  = (_Float16*)take((size_t)8 * 512 * DM * 2);
  _Float16*  xpjh  = (_Float16*)take((size_t)8 * 48 * DI * 2);
  _Float16*  outwh = (_Float16*)take((size_t)8 * DM * DI * 2);
  _Float16*  selfh = (_Float16*)take((size_t)DM * DM * 2);
  _Float16*  neigh = (_Float16*)take((size_t)DM * DM * 2);

  auto cvt = [&](const float* s, _Float16* d, int n) {
    k_f32_to_f16<<<(n + 255) / 256, 256, 0, stream>>>(s, d, n);
  };
  cvt(enc_w1, enc1h, DM * CIN);
  cvt(enc_w2, enc2h, DM * DM);
  cvt(m_in_w, inwh, 8 * 512 * DM);
  cvt(m_out_w, outwh, 8 * DM * DI);
  cvt(mp_self_w, selfh, DM * DM);
  cvt(mp_neig_w, neigh, DM * DM);
  k_xproj_pad<<<(8 * 48 * DI + 255) / 256, 256, 0, stream>>>(m_xproj_w, xpjh);

  k_encoder<<<SEQS, 256, 0, stream>>>(x, token_ids, enc1h, enc_b1, enc2h, enc_b2, hA);

  // two local bimamba layers (params 0,1 and 2,3), ping-pong hA/hB
  k_local_bimamba<<<SEQS, 256, 0, stream>>>(hA, hB, 0, inwh, m_conv_w, m_conv_b,
                                            xpjh, m_dt_w, m_dt_b, m_Alog, m_D, outwh);
  k_local_bimamba<<<SEQS, 256, 0, stream>>>(hB, hA, 2, inwh, m_conv_w, m_conv_b,
                                            xpjh, m_dt_w, m_dt_b, m_Alog, m_D, outwh);

  k_node_perm<<<(BB * NNODE * DM + 255) / 256, 256, 0, stream>>>(hA, perm, hgA);

  // two global bimamba layers (params 4,5 and 6,7), ping-pong hgA/hgB
  for (int gl = 0; gl < 2; ++gl) {
    const float* hin = gl ? (const float*)hgB : (const float*)hgA;
    float* hout = gl ? hgA : hgB;
    for (int dir = 0; dir < 2; ++dir) {
      int p = 4 + 2 * gl + dir;
      k_g_inproj<<<BB * 16, 256, 0, stream>>>(hin, xh_g, zh_g, p, dir, inwh);
      k_g_conv<<<(BB * GLN * DI + 255) / 256, 256, 0, stream>>>(xh_g, xch_g, p, m_conv_w, m_conv_b);
      k_g_xproj<<<BB * 16, 256, 0, stream>>>(xch_g, dt_g, Bm_g, Cm_g, p, xpjh, m_dt_w, m_dt_b);
      k_g_scan<<<BB, 256, 0, stream>>>(dt_g, xch_g, zh_g, Bm_g, Cm_g, yh_g, p, m_Alog, m_D);
      k_g_outproj<<<BB * 16, 256, 0, stream>>>(yh_g, hout, p, dir, outwh);
    }
  }

  k_invperm<<<(BB * NNODE * DM + 255) / 256, 256, 0, stream>>>(hgA, perm, feat, nei);
  k_edges<<<(2 * EE * BB * DM) / 256, 256, 0, stream>>>(feat, nei, edge_index);
  k_mpnn<<<(BB * NNODE) / 16, 256, 0, stream>>>(feat, nei, selfh, mp_self_b, neigh, mp_neig_b, out);
}
